// AUGRUCell_64596308132456
// MI455X (gfx1250) — compile-verified, matrix operand order checked by reading, not run on hardware
//
#include <hip/hip_runtime.h>
#include <hip/hip_bf16.h>

// AUGRU cell fused kernel for gfx1250 (MI455X), v2.
// B=16384, D=H=512. Six GEMMs fused into one pass with bf16 WMMA (f32 acc).
//
// v2: 4x M-tile reuse of weight fragments. Block = 256 thr (8 wave32) owns a
// 64-row x 128-col output tile; each wave owns 1 column tile (nt) and loops
// 4 row tiles, so each set of 6 B-fragments feeds 24 WMMAs. This cuts L2
// weight-fragment traffic from ~3GB to ~770MB and hides L2 latency.
//
// d_ws layout: packed bf16 weights, 6 matrices * 512*512 * 2B = 3,145,728 B.
// Packed B-fragment layout per matrix: [nt(32)][kt(16)][lane(32)][16 bf16]
//   lane (0..31): column n = nt*16 + (lane&15)
//   lanes 0-15 : elems 0-7 -> K = kt*32+0..7,  elems 8-15 -> K = kt*32+16..23
//   lanes 16-31: elems 0-7 -> K = kt*32+8..15, elems 8-15 -> K = kt*32+24..31
// (mirrors the CDNA5 ISA 16-bit A/B matrix VGPR striping, wave32)

typedef __attribute__((ext_vector_type(16))) __bf16 v16bf;
typedef __attribute__((ext_vector_type(8)))  float  v8f;

#define NB 16384
#define DK 512
#define NH 512
#define MT 4          // M tiles per wave (rows per block = MT*16 = 64)

// ---------------------------------------------------------------------------
// Kernel 1: repack + convert the six f32 weight matrices into WMMA-B-fragment
// order (bf16). One thread per (mat, nt, kt, lane) fragment slice: gathers 16
// strided f32, writes 32 contiguous bytes. 98304 threads total.
// ---------------------------------------------------------------------------
__global__ void augru_repack_weights(const float* __restrict__ Wu,
                                     const float* __restrict__ Uu,
                                     const float* __restrict__ Wr,
                                     const float* __restrict__ Ur,
                                     const float* __restrict__ Wh,
                                     const float* __restrict__ Uh,
                                     __bf16* __restrict__ packed) {
    int tid  = blockIdx.x * blockDim.x + threadIdx.x;   // 0 .. 98303
    int lane = tid & 31;
    int kt   = (tid >> 5) & 15;
    int nt   = (tid >> 9) & 31;
    int mat  = tid >> 14;                               // 0..5

    const float* srcs[6] = {Wu, Uu, Wr, Ur, Wh, Uh};
    const float* __restrict__ W = srcs[mat];

    int n     = nt * 16 + (lane & 15);
    int kbase = kt * 32 + ((lane >> 4) ? 8 : 0);

    __bf16* __restrict__ dst = packed + (size_t)tid * 16;
#pragma unroll
    for (int e = 0; e < 8; ++e)
        dst[e] = (__bf16)W[(size_t)(kbase + e) * NH + n];
#pragma unroll
    for (int e = 0; e < 8; ++e)
        dst[8 + e] = (__bf16)W[(size_t)(kbase + 16 + e) * NH + n];
}

// ---------------------------------------------------------------------------
// Kernel 2: fused AUGRU. Block = 256 threads (8 wave32), owns a 64-row x
// 128-col output tile. x/hidden 64x512 panels staged once in LDS as bf16
// (128 KB) and shared by all 8 waves. Each wave: 16 K-chunks; per chunk load
// 6 B-fragments once, reuse over 4 M tiles -> 24 WMMAs per chunk.
// ---------------------------------------------------------------------------
__global__ __launch_bounds__(256)
void augru_fused(const float* __restrict__ x,
                 const float* __restrict__ att,
                 const float* __restrict__ hidden,
                 const __bf16* __restrict__ packed,
                 const float* __restrict__ bu,
                 const float* __restrict__ br,
                 const float* __restrict__ bh,
                 float* __restrict__ out) {
    __shared__ __bf16 lx[64 * DK];   // 64 KB
    __shared__ __bf16 lh[64 * DK];   // 64 KB

    const int t        = threadIdx.x;
    const int row_base = blockIdx.x * (MT * 16);

    // ---- stage x / hidden row panels into LDS as bf16 (coalesced float4) ----
    // 64 rows * 128 float4 per matrix = 8192 float4; 32 per thread.
#pragma unroll 8
    for (int i = 0; i < 32; ++i) {
        int idx = t + i * 256;
        int row = idx >> 7;
        int kq  = (idx & 127) << 2;
        const float4 xv = *(const float4*)(x      + (size_t)(row_base + row) * DK + kq);
        const float4 hv = *(const float4*)(hidden + (size_t)(row_base + row) * DK + kq);
        __bf16* px = &lx[row * DK + kq];
        __bf16* ph = &lh[row * DK + kq];
        px[0] = (__bf16)xv.x; px[1] = (__bf16)xv.y; px[2] = (__bf16)xv.z; px[3] = (__bf16)xv.w;
        ph[0] = (__bf16)hv.x; ph[1] = (__bf16)hv.y; ph[2] = (__bf16)hv.z; ph[3] = (__bf16)hv.w;
    }
    __syncthreads();

    const int wave = t >> 5;
    const int lane = t & 31;
    const int nt   = blockIdx.y * 8 + wave;   // 0..31 (column tile)
    const int m    = lane & 15;
    const int hi   = lane >> 4;

    // Accumulators: u and r gates merge both GEMM products; candidate keeps
    // x@W_h and hidden@U_h separate (r scales only the latter).
    v8f acc_u[MT], acc_r[MT], acc_xh[MT], acc_hh[MT];
#pragma unroll
    for (int mt = 0; mt < MT; ++mt) {
        acc_u[mt]  = (v8f){};
        acc_r[mt]  = (v8f){};
        acc_xh[mt] = (v8f){};
        acc_hh[mt] = (v8f){};
    }

    const size_t matstride = (size_t)32 * 16 * 32 * 16;   // elems per matrix
    const size_t ntbase    = (size_t)nt * (16 * 32 * 16);
    const size_t lanebase  = (size_t)lane * 16;

    union Frag { v16bf v; uint4 q[2]; };

#pragma unroll 2
    for (int kt = 0; kt < 16; ++kt) {
        const int k0 = kt * 32;
        const int a0 = k0 + (hi ? 8 : 0);
        const int a1 = a0 + 16;

        // B fragments: contiguous 32B per lane, L2-resident packed weights.
        // Loaded once per chunk, reused by MT row tiles (24 WMMAs).
        const size_t fb = ntbase + (size_t)kt * (32 * 16) + lanebase;
        const v16bf bWu = *(const v16bf*)(packed + 0 * matstride + fb);
        const v16bf bUu = *(const v16bf*)(packed + 1 * matstride + fb);
        const v16bf bWr = *(const v16bf*)(packed + 2 * matstride + fb);
        const v16bf bUr = *(const v16bf*)(packed + 3 * matstride + fb);
        const v16bf bWh = *(const v16bf*)(packed + 4 * matstride + fb);
        const v16bf bUh = *(const v16bf*)(packed + 5 * matstride + fb);

        if (kt < 15) {
            __builtin_prefetch(packed + 0 * matstride + fb + (32 * 16), 0, 0);
            __builtin_prefetch(packed + 3 * matstride + fb + (32 * 16), 0, 0);
        }

#pragma unroll
        for (int mt = 0; mt < MT; ++mt) {
            const int arow = (mt * 16 + m) * DK;
            Frag ax, ah;
            ax.q[0] = *(const uint4*)&lx[arow + a0];
            ax.q[1] = *(const uint4*)&lx[arow + a1];
            ah.q[0] = *(const uint4*)&lh[arow + a0];
            ah.q[1] = *(const uint4*)&lh[arow + a1];

            acc_u[mt]  = __builtin_amdgcn_wmma_f32_16x16x32_bf16(false, ax.v, false, bWu, (short)0, acc_u[mt],  false, false);
            acc_r[mt]  = __builtin_amdgcn_wmma_f32_16x16x32_bf16(false, ax.v, false, bWr, (short)0, acc_r[mt],  false, false);
            acc_xh[mt] = __builtin_amdgcn_wmma_f32_16x16x32_bf16(false, ax.v, false, bWh, (short)0, acc_xh[mt], false, false);
            acc_u[mt]  = __builtin_amdgcn_wmma_f32_16x16x32_bf16(false, ah.v, false, bUu, (short)0, acc_u[mt],  false, false);
            acc_r[mt]  = __builtin_amdgcn_wmma_f32_16x16x32_bf16(false, ah.v, false, bUr, (short)0, acc_r[mt],  false, false);
            acc_hh[mt] = __builtin_amdgcn_wmma_f32_16x16x32_bf16(false, ah.v, false, bUh, (short)0, acc_hh[mt], false, false);
        }
    }

    // ---- epilogue: C/D layout -> VGPR r holds row (r + hi*8), col = lane&15 ----
    const int col = nt * 16 + m;
    const float bub = bu[col];
    const float brb = br[col];
    const float bhb = bh[col];

#pragma unroll
    for (int mt = 0; mt < MT; ++mt) {
#pragma unroll
        for (int r = 0; r < 8; ++r) {
            const int row = row_base + mt * 16 + r + (hi ? 8 : 0);
            const float a  = att[row];
            const float hp = hidden[(size_t)row * NH + col];

            const float u  = a / (1.0f + __expf(-(acc_u[mt][r] + bub)));    // att * sigmoid
            const float rr = 1.0f / (1.0f + __expf(-(acc_r[mt][r] + brb))); // reset gate
            const float pre = acc_xh[mt][r] + rr * acc_hh[mt][r] + bhb;
            const float e2  = __expf(-2.0f * pre);
            const float hh  = (1.0f - e2) / (1.0f + e2);                    // tanh

            out[(size_t)row * NH + col] = (1.0f - u) * hp + u * hh;
        }
    }
}

// ---------------------------------------------------------------------------
extern "C" void kernel_launch(void* const* d_in, const int* in_sizes, int n_in,
                              void* d_out, int out_size, void* d_ws, size_t ws_size,
                              hipStream_t stream) {
    const float* x    = (const float*)d_in[0];
    const float* att  = (const float*)d_in[1];
    const float* hid  = (const float*)d_in[2];
    const float* Wu   = (const float*)d_in[3];
    const float* Uu   = (const float*)d_in[4];
    const float* bu   = (const float*)d_in[5];
    const float* Wr   = (const float*)d_in[6];
    const float* Ur   = (const float*)d_in[7];
    const float* br   = (const float*)d_in[8];
    const float* Wh   = (const float*)d_in[9];
    const float* Uh   = (const float*)d_in[10];
    const float* bh   = (const float*)d_in[11];
    float* out        = (float*)d_out;

    __bf16* packed = (__bf16*)d_ws;   // needs 3,145,728 bytes

    // 6 mats * 32 nt * 16 kt * 32 lanes = 98304 threads
    augru_repack_weights<<<98304 / 256, 256, 0, stream>>>(Wu, Uu, Wr, Ur, Wh, Uh, packed);

    dim3 grid(NB / (MT * 16), NH / 128);   // (256, 4)
    augru_fused<<<grid, 256, 0, stream>>>(x, att, hid, packed, bu, br, bh, out);
}
